// New_msa_38860864094485
// MI455X (gfx1250) — matrix operand order repaired and checked by reading, not compiled, not run on hardware
//
#include <hip/hip_runtime.h>
#include <math.h>

// ---------------- constants ----------------
#define NB      4
#define HH      256
#define WW      256
#define CDIM    128
#define NPIX    65536          // HH*WW per batch
#define NTOT    262144         // NB*NPIX

// ---------------- bf16 helpers ----------------
typedef __attribute__((ext_vector_type(16))) __bf16 bf16x16;
typedef __attribute__((ext_vector_type(8)))  float  f32x8;

static __device__ __forceinline__ unsigned short f2bf(float f) {
  return __builtin_bit_cast(unsigned short, (__bf16)f);     // native v_cvt bf16
}
static __device__ __forceinline__ unsigned int pack_bf2(float lo, float hi) {
  return (unsigned int)f2bf(lo) | ((unsigned int)f2bf(hi) << 16);
}
static __device__ __forceinline__ float bf2f(unsigned short s) {
  return __uint_as_float(((unsigned int)s) << 16);
}
static __device__ __forceinline__ int rfl(int x) {          // force wave-uniform -> SGPR
  return __builtin_amdgcn_readfirstlane(x);
}

union Frag16 {                    // 16 bf16 = 8 VGPRs
  bf16x16 v;
  unsigned short u[16];
  uint4 q[2];
};

static __device__ __forceinline__ f32x8 wmma_bf16(const Frag16& a, const Frag16& b, f32x8 c) {
  return __builtin_amdgcn_wmma_f32_16x16x32_bf16(false, a.v, false, b.v, (short)0, c, false, false);
}

// A fragment (16x32 bf16) from an LDS tile [rows][ldk], per ISA layout table:
// lane L: m = L%16; kbase = (L<16)?0:8 ; elems 0..7 = K kbase..kbase+7, elems 8..15 = K 16+kbase..+7
static __device__ __forceinline__ Frag16 ldsA(const unsigned short* smem, int row0, int ldk, int k0) {
  int lane = threadIdx.x & 31;
  int m    = lane & 15;
  int kb   = (lane >> 4) * 8;
  const unsigned short* p = smem + (row0 + m) * ldk + k0;
  Frag16 f;
  f.q[0] = *(const uint4*)(p + kb);
  f.q[1] = *(const uint4*)(p + 16 + kb);
  return f;
}

// B fragment (32x16 bf16) from transposed-B storage BT[col][K] (row-major, bf16):
// lane L: col = col0 + L%16; kbase = (L<16)?0:8; elems 0..7 = K kbase..kbase+7, elems 8..15 = K 16+kbase..+7
static __device__ __forceinline__ Frag16 ldBT(const unsigned short* BT, int ldk, int col0, int k0) {
  int lane = threadIdx.x & 31;
  const unsigned short* p = BT + (size_t)(col0 + (lane & 15)) * ldk + k0 + ((lane >> 4) << 3);
  Frag16 f;
  f.q[0] = *(const uint4*)(p);
  f.q[1] = *(const uint4*)(p + 16);
  return f;
}

// Gram GEMM fragments from channel-major [ch][65536] bf16 storage
static __device__ __forceinline__ Frag16 ldGramA(const unsigned short* base, int nbase) {
  int lane = threadIdx.x & 31;
  int m    = lane & 15;
  int kb   = (lane >> 4) * 8;
  const unsigned short* p = base + ((size_t)m << 16) + nbase;
  Frag16 f;
  f.q[0] = *(const uint4*)(p + kb);
  f.q[1] = *(const uint4*)(p + 16 + kb);
  return f;
}
static __device__ __forceinline__ Frag16 ldGramB(const unsigned short* base, int nbase) {
  int lane = threadIdx.x & 31;
  const unsigned short* p = base + ((size_t)(lane & 15) << 16) + nbase + ((lane >> 4) << 3);
  Frag16 f;
  f.q[0] = *(const uint4*)(p);
  f.q[1] = *(const uint4*)(p + 16);
  return f;
}

// ================= K0: weight conversion =================
__global__ __launch_bounds__(256) void prep_kernel(const float* __restrict__ Wq,
                                                   const float* __restrict__ Wk,
                                                   const float* __restrict__ Wv,
                                                   const float* __restrict__ pw_w,
                                                   unsigned short* __restrict__ WqkvT,
                                                   unsigned short* __restrict__ pwbf) {
  int idx = blockIdx.x * 256 + threadIdx.x;
  if (idx < 384 * 128) {
    int col = idx >> 7, cin = idx & 127;
    float v;
    if      (col < 128) v = Wq[cin * 128 + col];
    else if (col < 256) v = Wk[cin * 128 + (col - 128)];
    else                v = Wv[cin * 128 + (col - 256)];
    WqkvT[col * 128 + cin] = f2bf(v);
  } else if (idx < 384 * 128 + 128 * 512) {
    int i2 = idx - 384 * 128;
    pwbf[i2] = f2bf(pw_w[i2]);
  }
}

// ================= K1: fused QKV projection GEMM =================
// block = 64 pixels; out cols 0..127->Qc(ch-major bf16), 128..255->Kc, 256..383->Vf32(row-major)
__global__ __launch_bounds__(256) void qkv_kernel(const float* __restrict__ x,
                                                  const unsigned short* __restrict__ WqkvT,
                                                  unsigned short* __restrict__ Qc,
                                                  unsigned short* __restrict__ Kc,
                                                  float* __restrict__ Vf) {
  __shared__ __align__(16) unsigned short xs[64 * 128];
  int t = threadIdx.x;
  size_t gp0 = (size_t)blockIdx.x * 64;
  const float4* xsrc = (const float4*)(x + gp0 * 128);
#pragma unroll
  for (int i = 0; i < 8; ++i) {
    int e = t + i * 256;
    float4 v = xsrc[e];
    uint2 u; u.x = pack_bf2(v.x, v.y); u.y = pack_bf2(v.z, v.w);
    ((uint2*)xs)[e] = u;
  }
  __syncthreads();
  int wv = t >> 5, lane = t & 31;
  int b0   = (int)(gp0 >> 16);            // 64-pixel tile never crosses a batch
  int pix0 = (int)(gp0 & 65535);
  for (int tile = wv; tile < 96; tile += 8) {     // 4 px-subtiles x 24 col-tiles
    int pxs = rfl(tile / 24), ct = rfl(tile % 24);
    f32x8 acc; for (int r = 0; r < 8; ++r) acc[r] = 0.f;
#pragma unroll
    for (int ks = 0; ks < 4; ++ks) {
      Frag16 a = ldsA(xs, pxs * 16, 128, ks * 32);
      Frag16 b = ldBT(WqkvT, 128, ct * 16, ks * 32);
      acc = wmma_bf16(a, b, acc);
    }
    int colb = ct * 16 + (lane & 15);
    int prow = pix0 + pxs * 16 + ((lane >> 4) << 3);
    if (ct < 16) {                               // scalar-uniform routing
      unsigned short* dst = (ct < 8) ? Qc : Kc;
      int ch = colb & 127;
      size_t base = ((size_t)(b0 * 128 + ch) << 16) + prow;
#pragma unroll
      for (int r = 0; r < 8; ++r) dst[base + r] = f2bf(acc[r]);
    } else {
      int c = colb - 256;
      size_t base = (gp0 + pxs * 16 + ((lane >> 4) << 3)) * 128 + c;
#pragma unroll
      for (int r = 0; r < 8; ++r) Vf[base + (size_t)r * 128] = acc[r];
    }
  }
}

// ================= K2: per-channel inverse L2 norms =================
__global__ __launch_bounds__(256) void norm_kernel(const unsigned short* __restrict__ Qc,
                                                   const unsigned short* __restrict__ Kc,
                                                   float* __restrict__ invQ,
                                                   float* __restrict__ invK) {
  __shared__ float red[256];
  int t = threadIdx.x;
  int bid = blockIdx.x;              // 0..1023 : [Q 512 | K 512]
  int sel = bid >> 9, rem = bid & 511;
  const unsigned short* base = (sel ? Kc : Qc) + ((size_t)rem << 16);
  const uint2* p = (const uint2*)base;
  float s = 0.f;
  for (int i = t; i < 16384; i += 256) {
    uint2 u = p[i];
    float a0 = bf2f((unsigned short)(u.x & 0xffffu));
    float a1 = bf2f((unsigned short)(u.x >> 16));
    float a2 = bf2f((unsigned short)(u.y & 0xffffu));
    float a3 = bf2f((unsigned short)(u.y >> 16));
    s += a0 * a0 + a1 * a1 + a2 * a2 + a3 * a3;
  }
  red[t] = s; __syncthreads();
  for (int st = 128; st > 0; st >>= 1) { if (t < st) red[t] += red[t + st]; __syncthreads(); }
  if (t == 0) (sel ? invK : invQ)[rem] = 1.f / fmaxf(sqrtf(red[0]), 1e-12f);
}

// ================= K3: Gram(K=65536) via WMMA + softmax + fold A^T*Wproj =================
__global__ __launch_bounds__(256) void attn_kernel(const unsigned short* __restrict__ Qc,
                                                   const unsigned short* __restrict__ Kc,
                                                   const float* __restrict__ invQ,
                                                   const float* __restrict__ invK,
                                                   const float* __restrict__ rescale,
                                                   const float* __restrict__ Wproj,
                                                   unsigned short* __restrict__ MTbf) {
  __shared__ float S2[2048];
  __shared__ float Sf[1024];
  __shared__ float As[1024];
  int t = threadIdx.x, lane = t & 31, wv = t >> 5;
  int b_ = blockIdx.x >> 2, h = blockIdx.x & 3;
  int chbase = b_ * 128 + h * 32;
  int dtile = rfl((wv >> 1) & 1), etile = rfl(wv & 1), khs = rfl(wv >> 2);
  const unsigned short* Ab = Kc + ((size_t)(chbase + dtile * 16) << 16);
  const unsigned short* Bb = Qc + ((size_t)(chbase + etile * 16) << 16);
  f32x8 acc; for (int r = 0; r < 8; ++r) acc[r] = 0.f;
  int n0 = khs * 32768;
  for (int i = 0; i < 1024; ++i) {
    int nb = n0 + i * 32;
    __builtin_prefetch(Ab + nb + 4096, 0, 1);    // global_prefetch_b8, speculative OK
    __builtin_prefetch(Bb + nb + 4096, 0, 1);
    Frag16 a = ldGramA(Ab, nb);
    Frag16 b = ldGramB(Bb, nb);
    acc = wmma_bf16(a, b, acc);
  }
  {
    int e  = etile * 16 + (lane & 15);
    int d0 = dtile * 16 + (lane >> 4) * 8;
    for (int r = 0; r < 8; ++r) S2[khs * 1024 + (d0 + r) * 32 + e] = acc[r];
  }
  __syncthreads();
  for (int idx = t; idx < 1024; idx += 256) {
    int d = idx >> 5, e = idx & 31;
    Sf[idx] = (S2[idx] + S2[1024 + idx]) * invK[chbase + d] * invQ[chbase + e] * rescale[h];
  }
  __syncthreads();
  if (wv == 0) {                                  // 32 lanes: softmax row d over e
    int d = lane;
    float mx = -1e30f;
    for (int e = 0; e < 32; ++e) mx = fmaxf(mx, Sf[d * 32 + e]);
    float sm = 0.f;
    for (int e = 0; e < 32; ++e) { float ex = expf(Sf[d * 32 + e] - mx); As[d * 32 + e] = ex; sm += ex; }
    float inv = 1.f / sm;
    for (int e = 0; e < 32; ++e) As[d * 32 + e] *= inv;
  }
  __syncthreads();
  // M[h*32+e][o] = sum_d attn[d,e] * Wproj[h*32+d][o]; stored transposed: MT[o][h*32+e]
  for (int idx = t; idx < 4096; idx += 256) {
    int o = idx & 127, e = idx >> 7;
    float sum = 0.f;
    for (int d = 0; d < 32; ++d) sum += As[d * 32 + e] * Wproj[(size_t)(h * 32 + d) * 128 + o];
    MTbf[((size_t)b_ * 128 + o) * 128 + (h * 32 + e)] = f2bf(sum);
  }
}

// ================= K4: out_c = V * M_b + bproj =================
__global__ __launch_bounds__(256) void outc_kernel(const float* __restrict__ Vf,
                                                   const unsigned short* __restrict__ MTbf,
                                                   const float* __restrict__ bproj,
                                                   float* __restrict__ out) {
  __shared__ __align__(16) unsigned short vs[64 * 128];
  int t = threadIdx.x;
  size_t gp0 = (size_t)blockIdx.x * 64;
  const float4* src = (const float4*)(Vf + gp0 * 128);
#pragma unroll
  for (int i = 0; i < 8; ++i) {
    int e = t + i * 256;
    float4 v = src[e];
    uint2 u; u.x = pack_bf2(v.x, v.y); u.y = pack_bf2(v.z, v.w);
    ((uint2*)vs)[e] = u;
  }
  __syncthreads();
  size_t b_ = gp0 >> 16;
  const unsigned short* MT = MTbf + b_ * 128 * 128;
  int wv = t >> 5, lane = t & 31;
  for (int tile = wv; tile < 32; tile += 8) {     // 4 px-subtiles x 8 col-tiles
    int pxs = rfl(tile >> 3), ct = rfl(tile & 7);
    f32x8 acc; for (int r = 0; r < 8; ++r) acc[r] = 0.f;
#pragma unroll
    for (int ks = 0; ks < 4; ++ks) {
      Frag16 a = ldsA(vs, pxs * 16, 128, ks * 32);
      Frag16 b = ldBT(MT, 128, ct * 16, ks * 32);
      acc = wmma_bf16(a, b, acc);
    }
    int gcol = ct * 16 + (lane & 15);
    size_t prow0 = gp0 + pxs * 16 + (size_t)(lane >> 4) * 8;
    float bias = bproj[gcol];
#pragma unroll
    for (int r = 0; r < 8; ++r) out[(prow0 + r) * 128 + gcol] = acc[r] + bias;
  }
}

// ================= K5a: vertical 7-tap (gx shared by all 4 scales) =================
__global__ __launch_bounds__(256) void vconv_kernel(const float* __restrict__ Vf,
                                                    const float* __restrict__ dw_w,
                                                    float* __restrict__ tmp) {
  int idx = blockIdx.x * 256 + threadIdx.x;       // 8,388,608 float4 elements
  int c4 = idx & 31;
  int r  = idx >> 5;
  int x  = r & 255; int r2 = r >> 8; int y = r2 & 255; int b_ = r2 >> 8;
  float gx[7];
#pragma unroll
  for (int a = 0; a < 7; ++a) gx[a] = dw_w[a * 7 + 4];   // k[0][a][4] = exp(-coords[a]^2/2)
  float4 acc = make_float4(0.f, 0.f, 0.f, 0.f);
  const float4* src = (const float4*)Vf;
  size_t baserow = ((size_t)b_ << 16);
#pragma unroll
  for (int a = 0; a < 7; ++a) {
    int yy = y + a - 3;
    if (yy >= 0 && yy < 256) {
      float4 v = src[(baserow + (size_t)yy * 256 + x) * 32 + c4];
      acc.x += gx[a] * v.x; acc.y += gx[a] * v.y; acc.z += gx[a] * v.z; acc.w += gx[a] * v.w;
    }
  }
  ((float4*)tmp)[(baserow + (size_t)y * 256 + x) * 32 + c4] = acc;
}

// ================= K5b: horizontal 4-scale conv + GELU + pointwise GEMM (+= d_out) =====
__global__ __launch_bounds__(256) void gabor_pw_kernel(const float* __restrict__ tmp,
                                                       const float* __restrict__ dw_w,
                                                       const float* __restrict__ dw_b,
                                                       const unsigned short* __restrict__ pwbf,
                                                       const float* __restrict__ pw_b,
                                                       float* __restrict__ out) {
  __shared__ __align__(16) float ts[70 * 128];            // 64-px x-strip + halo(3), fp32
  __shared__ __align__(16) unsigned short gs[64 * 128];   // quarter of g, bf16
  int t = threadIdx.x;
  size_t gp0 = (size_t)blockIdx.x * 64;
  int b_ = (int)(gp0 >> 16); int rem = (int)(gp0 & 65535); int y = rem >> 8; int x0 = rem & 255;
  // stage tmp strip: async DMA global->LDS (ASYNCcnt path) for in-range 16B chunks,
  // plain LDS zero-stores (DScnt, covered by barrier) for the padded border.
  for (int i = t; i < 2240; i += 256) {          // 2240 x 16B = 70*128 floats
    int fo = i * 4;
    int px = fo >> 7, c = fo & 127;
    int xx = x0 - 3 + px;
    if (xx >= 0 && xx < 256) {
      const float* gp = tmp + (((size_t)b_ << 16) + (size_t)y * 256 + xx) * 128 + c;
      unsigned lds_addr = (unsigned)(uintptr_t)(&ts[fo]);           // low 32b of generic = LDS offset
      unsigned long long ga = (unsigned long long)(uintptr_t)gp;
      asm volatile("global_load_async_to_lds_b128 %0, %1, off"
                   :: "v"(lds_addr), "v"(ga) : "memory");
    } else {
      float4 z = make_float4(0.f, 0.f, 0.f, 0.f);
      *(float4*)(&ts[fo]) = z;
    }
  }
  asm volatile("s_wait_asynccnt 0x0" ::: "memory");
  float gy[4][7];
#pragma unroll
  for (int j = 0; j < 4; ++j)
#pragma unroll
    for (int a = 0; a < 7; ++a) gy[j][a] = dw_w[j * 49 + 28 + a];   // k[j][4][a]
  int wv = t >> 5, lane = t & 31;
  f32x8 acc4[4];
#pragma unroll
  for (int tt = 0; tt < 4; ++tt) for (int r = 0; r < 8; ++r) acc4[tt][r] = 0.f;
  for (int q = 0; q < 4; ++q) {                   // K-dim quarters of the 512-wide g
    __syncthreads();
    for (int i = t; i < 8192; i += 256) {
      int px = i >> 7, local = i & 127;
      int cc = q * 128 + local;                   // g channel = 4c + j
      int c = cc >> 2, j = cc & 3;
      float conv = 0.f;
#pragma unroll
      for (int a = 0; a < 7; ++a) conv += gy[j][a] * ts[(px + a) * 128 + c];
      float val = conv + dw_b[cc];
      float g = 0.5f * val * (1.f + erff(val * 0.70710678118f));   // exact GELU
      gs[px * 128 + local] = f2bf(g);
    }
    __syncthreads();
#pragma unroll
    for (int tt = 0; tt < 4; ++tt) {              // wave wv owns col-tile wv, px-subtile tt
      int pxs = tt, ct = wv;
#pragma unroll
      for (int ks = 0; ks < 4; ++ks) {
        Frag16 a = ldsA(gs, pxs * 16, 128, ks * 32);
        Frag16 b = ldBT(pwbf, 512, ct * 16, q * 128 + ks * 32);
        acc4[tt] = wmma_bf16(a, b, acc4[tt]);
      }
    }
  }
  int gcol = wv * 16 + (lane & 15);
  float bias = pw_b[gcol];
#pragma unroll
  for (int tt = 0; tt < 4; ++tt) {
    size_t prow0 = gp0 + tt * 16 + (size_t)(lane >> 4) * 8;
#pragma unroll
    for (int r = 0; r < 8; ++r) {
      size_t o = (prow0 + r) * 128 + gcol;
      out[o] = out[o] + acc4[tt][r] + bias;
    }
  }
}

// ================= launcher =================
extern "C" void kernel_launch(void* const* d_in, const int* in_sizes, int n_in,
                              void* d_out, int out_size, void* d_ws, size_t ws_size,
                              hipStream_t stream) {
  (void)in_sizes; (void)n_in; (void)out_size; (void)ws_size;
  const float* x_in    = (const float*)d_in[0];
  const float* Wq      = (const float*)d_in[1];
  const float* Wk      = (const float*)d_in[2];
  const float* Wv      = (const float*)d_in[3];
  const float* rescale = (const float*)d_in[4];
  const float* Wproj   = (const float*)d_in[5];
  const float* bproj   = (const float*)d_in[6];
  const float* dw_w    = (const float*)d_in[7];
  const float* dw_b    = (const float*)d_in[8];
  const float* pw_w    = (const float*)d_in[9];
  const float* pw_b    = (const float*)d_in[10];
  float* out = (float*)d_out;

  // workspace layout (≈269 MB; tmp aliases Qc∪Kc which are consumed before K5a)
  char* ws = (char*)d_ws;
  float*          Vf    = (float*)(ws);                                   // 134,217,728 B
  unsigned short* Qc    = (unsigned short*)(ws + 134217728ull);           //  67,108,864 B
  unsigned short* Kc    = (unsigned short*)(ws + 201326592ull);           //  67,108,864 B
  float*          tmp   = (float*)(ws + 134217728ull);                    // alias (134 MB)
  char* sm = ws + 268435456ull;
  float*          invQ  = (float*)sm;                 sm += 2048;
  float*          invK  = (float*)sm;                 sm += 2048;
  unsigned short* WqkvT = (unsigned short*)sm;        sm += 98304;
  unsigned short* pwbf  = (unsigned short*)sm;        sm += 131072;
  unsigned short* MTbf  = (unsigned short*)sm;        sm += 131072;

  hipLaunchKernelGGL(prep_kernel,     dim3(448),   dim3(256), 0, stream, Wq, Wk, Wv, pw_w, WqkvT, pwbf);
  hipLaunchKernelGGL(qkv_kernel,      dim3(4096),  dim3(256), 0, stream, x_in, WqkvT, Qc, Kc, Vf);
  hipLaunchKernelGGL(norm_kernel,     dim3(1024),  dim3(256), 0, stream, Qc, Kc, invQ, invK);
  hipLaunchKernelGGL(attn_kernel,     dim3(16),    dim3(256), 0, stream, Qc, Kc, invQ, invK, rescale, Wproj, MTbf);
  hipLaunchKernelGGL(outc_kernel,     dim3(4096),  dim3(256), 0, stream, Vf, MTbf, bproj, out);
  hipLaunchKernelGGL(vconv_kernel,    dim3(32768), dim3(256), 0, stream, Vf, dw_w, tmp);
  hipLaunchKernelGGL(gabor_pw_kernel, dim3(4096),  dim3(256), 0, stream, tmp, dw_w, dw_b, pwbf, pw_b, out);
}
